// GAT_9070970929361
// MI455X (gfx1250) — compile-verified, matrix operand order checked by reading, not run on hardware
//
#include <hip/hip_runtime.h>
#include <hip/hip_bf16.h>

typedef __attribute__((ext_vector_type(16))) _Float16 v16h;
typedef __attribute__((ext_vector_type(8)))  float    v8f;

#define IN_DIM    192
#define KSTEPS    6          // 192 / 32
#define NEG_SLOPE 0.2f

// K index inside a 32-wide k-step for the 16-bit WMMA A/B register layout.
// slot = 2*vgpr + half (0..15).  Lanes 0-15: K = {0..7, 16..23}; lanes 16-31: K = {8..15, 24..31}.
__device__ __forceinline__ int kmap(int lane, int slot) {
    int h  = slot & 1;
    int v  = slot >> 1;
    int g  = v >> 2;        // 0: first K-octet group, 1: second
    int vi = v & 3;
    int base = g ? ((lane < 16) ? 16 : 24) : ((lane < 16) ? 0 : 8);
    return base + 2 * vi + h;
}

// ---------- pass 0: zero accumulators ----------
__global__ void gat_init(unsigned* __restrict__ maxkey, float* __restrict__ sum,
                         float* __restrict__ out, int N) {
    int i = blockIdx.x * blockDim.x + threadIdx.x;
    if (i < N) {
        maxkey[i]      = 0u;     // encodes -NaN: below every real key
        sum[i]         = 0.f;
        out[2 * i]     = 0.f;
        out[2 * i + 1] = 0.f;
    }
}

// ---------- pass 1: build fused B operand  W4 = [W | W@att_src | W@att_dst | 0...] ----------
// Pre-packed into the wave32 16-bit B register layout: packedB[(kstep*32 + lane)*16 + slot].
__global__ void gat_prep_B(const float* __restrict__ W,
                           const float* __restrict__ att_src,
                           const float* __restrict__ att_dst,
                           _Float16* __restrict__ packedB) {
    int t = blockIdx.x * blockDim.x + threadIdx.x;      // 0 .. 6*32-1
    if (t >= KSTEPS * 32) return;
    int kstep = t >> 5;
    int lane  = t & 31;
    int n     = lane & 15;                              // output column this lane feeds
    float as0 = att_src[0], as1 = att_src[1];
    float ad0 = att_dst[0], ad1 = att_dst[1];
    for (int slot = 0; slot < 16; ++slot) {
        int   K  = kstep * 32 + kmap(lane, slot);
        float w0 = W[K * 2 + 0], w1 = W[K * 2 + 1];
        float val = 0.f;
        if      (n == 0) val = w0;
        else if (n == 1) val = w1;
        else if (n == 2) val = w0 * as0 + w1 * as1;     // a_src column
        else if (n == 3) val = w0 * ad0 + w1 * ad1;     // a_dst column
        packedB[(kstep * 32 + lane) * 16 + slot] = (_Float16)val;
    }
}

// ---------- pass 2: WMMA projection  xproj4[N][4] = {proj0, proj1, a_src, a_dst} ----------
__global__ void __launch_bounds__(128)
gat_proj_wmma(const float* __restrict__ x,
              const _Float16* __restrict__ packedB,
              float* __restrict__ xproj4,
              int nTiles, int N) {
    int wave = (blockIdx.x * blockDim.x + threadIdx.x) >> 5;
    int lane = threadIdx.x & 31;
    if (wave >= nTiles) return;
    int m0  = wave * 16;
    int row = m0 + (lane & 15);
    if (row >= N) row = N - 1;                          // clamp (reads only)

    const v16h* pB = (const v16h*)packedB;
    v8f c = {};
    #pragma unroll
    for (int ks = 0; ks < KSTEPS; ++ks) {
        int kb = ks * 32;
        v16h a;
        // group 0: 8 contiguous f32 at K = kb + (lane<16 ? 0 : 8)
        {
            const float4* p = (const float4*)(x + (size_t)row * IN_DIM + kb + ((lane < 16) ? 0 : 8));
            float4 f0 = p[0], f1 = p[1];
            a[0] = (_Float16)f0.x; a[1] = (_Float16)f0.y; a[2] = (_Float16)f0.z; a[3] = (_Float16)f0.w;
            a[4] = (_Float16)f1.x; a[5] = (_Float16)f1.y; a[6] = (_Float16)f1.z; a[7] = (_Float16)f1.w;
        }
        // group 1: 8 contiguous f32 at K = kb + (lane<16 ? 16 : 24)
        {
            const float4* p = (const float4*)(x + (size_t)row * IN_DIM + kb + ((lane < 16) ? 16 : 24));
            float4 f0 = p[0], f1 = p[1];
            a[8]  = (_Float16)f0.x; a[9]  = (_Float16)f0.y; a[10] = (_Float16)f0.z; a[11] = (_Float16)f0.w;
            a[12] = (_Float16)f1.x; a[13] = (_Float16)f1.y; a[14] = (_Float16)f1.z; a[15] = (_Float16)f1.w;
        }
        v16h b = pB[ks * 32 + lane];
        c = __builtin_amdgcn_wmma_f32_16x16x32_f16(false, a, false, b, (short)0, c,
                                                   false, false);
    }
    // C/D layout: VGPR v -> M = v (lanes 0-15) or 8+v (lanes 16-31); column = lane & 15
    int n = lane & 15;
    if (n < 4) {
        int mbase = m0 + ((lane < 16) ? 0 : 8);
        #pragma unroll
        for (int v = 0; v < 8; ++v) {
            int m = mbase + v;
            if (m < N) xproj4[(size_t)m * 4 + n] = c[v];
        }
    }
}

// ---------- monotonic float<->uint keys for atomic segment-max ----------
__device__ __forceinline__ unsigned fkey(float f) {
    unsigned u = __float_as_uint(f);
    return (u & 0x80000000u) ? ~u : (u | 0x80000000u);
}
__device__ __forceinline__ float funkey(unsigned k) {
    unsigned u = (k & 0x80000000u) ? (k & 0x7FFFFFFFu) : ~k;
    return __uint_as_float(u);
}

// ---------- pass 3: segment max of leaky-relu attention logits ----------
__global__ void gat_edge_max(const int* __restrict__ ei, int E, int N,
                             const float* __restrict__ xproj4,
                             unsigned* __restrict__ maxkey) {
    long long t = (long long)blockIdx.x * blockDim.x + threadIdx.x;
    if (t >= (long long)E + N) return;
    int s, d;
    if (t < E) { s = ei[t]; d = ei[(long long)E + t]; }
    else       { s = d = (int)(t - E); }                 // self loop
    float a = xproj4[(size_t)s * 4 + 2] + xproj4[(size_t)d * 4 + 3];
    a = (a >= 0.f) ? a : NEG_SLOPE * a;
    atomicMax(&maxkey[d], fkey(a));
}

// ---------- pass 4: fused exp / denom-sum / weighted scatter (unnormalized) ----------
__global__ void gat_edge_acc(const int* __restrict__ ei, int E, int N,
                             const float* __restrict__ xproj4,
                             const unsigned* __restrict__ maxkey,
                             float* __restrict__ sum, float* __restrict__ out) {
    long long t = (long long)blockIdx.x * blockDim.x + threadIdx.x;
    if (t >= (long long)E + N) return;
    int s, d;
    if (t < E) { s = ei[t]; d = ei[(long long)E + t]; }
    else       { s = d = (int)(t - E); }
    float a = xproj4[(size_t)s * 4 + 2] + xproj4[(size_t)d * 4 + 3];
    a = (a >= 0.f) ? a : NEG_SLOPE * a;
    float p = __expf(a - funkey(maxkey[d]));
    atomicAdd(&sum[d], p);
    atomicAdd(&out[2 * (size_t)d + 0], p * xproj4[(size_t)s * 4 + 0]);
    atomicAdd(&out[2 * (size_t)d + 1], p * xproj4[(size_t)s * 4 + 1]);
}

// ---------- pass 5: normalize + bias ----------
__global__ void gat_final(float* __restrict__ out, const float* __restrict__ sum,
                          const float* __restrict__ bias, int N) {
    int i = blockIdx.x * blockDim.x + threadIdx.x;
    if (i < N) {
        float inv = 1.0f / (sum[i] + 1e-16f);
        out[2 * i]     = out[2 * i]     * inv + bias[0];
        out[2 * i + 1] = out[2 * i + 1] * inv + bias[1];
    }
}

extern "C" void kernel_launch(void* const* d_in, const int* in_sizes, int n_in,
                              void* d_out, int out_size, void* d_ws, size_t ws_size,
                              hipStream_t stream) {
    // inputs: x, edge_index, edge_attr(unused), W, att_src, att_dst, bias
    const float* x       = (const float*)d_in[0];
    const int*   ei      = (const int*)d_in[1];
    const float* W       = (const float*)d_in[3];
    const float* att_src = (const float*)d_in[4];
    const float* att_dst = (const float*)d_in[5];
    const float* bias    = (const float*)d_in[6];
    float* out = (float*)d_out;

    const int N = in_sizes[0] / IN_DIM;
    const int E = in_sizes[1] / 2;

    // workspace layout
    char* ws = (char*)d_ws;
    _Float16* packedB = (_Float16*)ws;                               // 6*32*16*2 = 6144 B
    float*    xproj4  = (float*)(ws + 8192);                          // N*4 f32
    unsigned* maxkey  = (unsigned*)(ws + 8192 + (size_t)N * 4 * 4);   // N u32
    float*    sum     = (float*)((char*)maxkey + (size_t)N * 4);      // N f32

    const long long totE = (long long)E + N;

    gat_init<<<(N + 255) / 256, 256, 0, stream>>>(maxkey, sum, out, N);
    gat_prep_B<<<1, KSTEPS * 32, 0, stream>>>(W, att_src, att_dst, packedB);

    const int nTiles = (N + 15) / 16;                 // one wave per 16-row tile
    const int wavesPerBlock = 4;                      // 128 threads
    gat_proj_wmma<<<(nTiles + wavesPerBlock - 1) / wavesPerBlock, wavesPerBlock * 32, 0, stream>>>(
        x, packedB, xproj4, nTiles, N);

    const int eb = 256;
    gat_edge_max<<<(int)((totE + eb - 1) / eb), eb, 0, stream>>>(ei, E, N, xproj4, maxkey);
    gat_edge_acc<<<(int)((totE + eb - 1) / eb), eb, 0, stream>>>(ei, E, N, xproj4, maxkey, sum, out);
    gat_final<<<(N + 255) / 256, 256, 0, stream>>>(out, sum, bias, N);
}
